// MultiHeadAttention_7413113553038
// MI455X (gfx1250) — matrix operand order repaired and checked by reading, not compiled, not run on hardware
//
#include <hip/hip_runtime.h>
#include <hip/hip_bf16.h>
#include <stdint.h>

// ---------------------------------------------------------------------------
// MI455X (gfx1250) multi-head attention with Gaussian prior.
// bf16 WMMA everywhere + async global->LDS double-buffered K/V staging.
// ---------------------------------------------------------------------------

typedef __bf16 bf16_t;
typedef __attribute__((ext_vector_type(8)))  __bf16 v8bf;
typedef __attribute__((ext_vector_type(16))) __bf16 v16bf;
typedef __attribute__((ext_vector_type(8)))  float  v8f;

#define F_SIZE 256
#define SEQ    2048
#define NROWS  4096   /* B * S */
#define NHEADS 4

__device__ __forceinline__ v16bf bf_cat(v8bf lo, v8bf hi) {
  return __builtin_shufflevector(lo, hi, 0,1,2,3,4,5,6,7,8,9,10,11,12,13,14,15);
}

__device__ __forceinline__ v8f wmma_bf16(v16bf a, v16bf b, v8f c) {
  // D = A(16x32 bf16) * B(32x16 bf16) + C(16x16 f32)
  return __builtin_amdgcn_wmma_f32_16x16x32_bf16(false, a, false, b, (short)0, c,
                                                 false, false);
}

// A operand (16x32, row-major source). Lane m = L&15, half = L>>4;
// elements 0..7 = K[half*8 ..], 8..15 = K[16+half*8 ..].
__device__ __forceinline__ v16bf load_a16x32(const bf16_t* __restrict__ X, int ld,
                                             int row0, int k0, int lane) {
  const int m = lane & 15, half = lane >> 4;
  const bf16_t* r = X + (size_t)(row0 + m) * ld + k0 + half * 8;
  v8bf lo = *reinterpret_cast<const v8bf*>(r);
  v8bf hi = *reinterpret_cast<const v8bf*>(r + 16);
  return bf_cat(lo, hi);
}

// B operand (32x16) from "transposed" storage T[n][k] (n-major, k contiguous).
// Lane n = L&15, kh = L>>4; elements 0..7 = K[kh*8 ..], 8..15 = K[16+kh*8 ..].
__device__ __forceinline__ v16bf load_b32x16(const bf16_t* __restrict__ T, int ld,
                                             int n0, int k0, int lane) {
  const int n = lane & 15, kh = lane >> 4;
  const bf16_t* r = T + (size_t)(n0 + n) * ld + k0 + kh * 8;
  v8bf lo = *reinterpret_cast<const v8bf*>(r);
  v8bf hi = *reinterpret_cast<const v8bf*>(r + 16);
  return bf_cat(lo, hi);
}

// Async 16B global -> LDS copy (per lane). lds_off is a wave-relative LDS byte
// address (low 32 bits of the flat pointer == LDS offset per aperture rules).
__device__ __forceinline__ void async_copy16(uint32_t lds_off, const void* src) {
  asm volatile("global_load_async_to_lds_b128 %0, %1, off"
               :: "v"(lds_off), "v"(src) : "memory");
}

// ---------------------------------------------------------------------------
// Kernel 1a: convert fp32 inputs -> bf16
__global__ void cvt_inputs_kernel(const float* __restrict__ xq,
                                  const float* __restrict__ xk,
                                  const float* __restrict__ xv,
                                  bf16_t* __restrict__ xqb,
                                  bf16_t* __restrict__ xkb,
                                  bf16_t* __restrict__ xvb, int n) {
  int i = blockIdx.x * blockDim.x + threadIdx.x;
  if (i < n) {
    xqb[i] = (bf16_t)xq[i];
    xkb[i] = (bf16_t)xk[i];
    xvb[i] = (bf16_t)xv[i];
  }
}

// Kernel 1b: convert + transpose weight W[in][out] -> WT[out][in], bf16
__global__ void cvt_wt_kernel(const float* __restrict__ w, bf16_t* __restrict__ wt,
                              int rows, int cols) {
  int i = blockIdx.x * blockDim.x + threadIdx.x;
  if (i < rows * cols) {
    int r = i / cols, c = i % cols;
    wt[(size_t)c * rows + r] = (bf16_t)w[i];
  }
}

// ---------------------------------------------------------------------------
// Kernel 2: QKV projections. X[4096,256] @ W[256,256]. One 16x16 tile/wave.
__global__ void proj_kernel(const bf16_t* __restrict__ xq,
                            const bf16_t* __restrict__ xk,
                            const bf16_t* __restrict__ xv,
                            const bf16_t* __restrict__ wqT,
                            const bf16_t* __restrict__ wkT,
                            const bf16_t* __restrict__ wvT,
                            bf16_t* __restrict__ qp,
                            bf16_t* __restrict__ kp,
                            bf16_t* __restrict__ vt) {
  const int lane = threadIdx.x & 31;
  const int wave = threadIdx.x >> 5;
  const int mat  = blockIdx.y;
  const bf16_t* X  = (mat == 0) ? xq : (mat == 1) ? xk : xv;
  const bf16_t* WT = (mat == 0) ? wqT : (mat == 1) ? wkT : wvT;

  const int tile = blockIdx.x * 8 + wave;     // 0..4095
  const int row0 = (tile >> 4) * 16;
  const int col0 = (tile & 15) * 16;

  v8f acc = {0.f, 0.f, 0.f, 0.f, 0.f, 0.f, 0.f, 0.f};
#pragma unroll
  for (int kc = 0; kc < 8; ++kc) {
    v16bf a = load_a16x32(X, F_SIZE, row0, kc * 32, lane);
    v16bf b = load_b32x16(WT, F_SIZE, col0, kc * 32, lane);
    acc = wmma_bf16(a, b, acc);
  }

  const int n = lane & 15, half = lane >> 4;
  if (mat < 2) {
    bf16_t* Y = (mat == 0) ? qp : kp;
#pragma unroll
    for (int v = 0; v < 8; ++v) {
      int r = row0 + v + 8 * half;
      Y[(size_t)r * F_SIZE + col0 + n] = (bf16_t)acc[v];
    }
  } else {
    // Vt[b][h][feat][token]
#pragma unroll
    for (int v = 0; v < 8; ++v) {
      int r = row0 + v + 8 * half;
      int b = r >> 11, s = r & (SEQ - 1);
      int col = col0 + n;
      int h = col >> 6, f = col & 63;
      vt[(((size_t)b * NHEADS + h) * 64 + f) * SEQ + s] = (bf16_t)acc[v];
    }
  }
}

// ---------------------------------------------------------------------------
// Kernel 3: flash attention. Block = 8 waves sharing one (b,h); each wave owns
// a 16-query tile. K/V tiles double-buffered in LDS via async copies.
__global__ void __launch_bounds__(256) attn_kernel(const bf16_t* __restrict__ qp,
                                                   const bf16_t* __restrict__ kp,
                                                   const bf16_t* __restrict__ vt,
                                                   bf16_t* __restrict__ ctx) {
  // [buf][ K tile 32x64 (4KB) | Vt tile 64x32 (4KB) ]
  __shared__ bf16_t kvbuf[2][4096];
  __shared__ bf16_t pbuf[8][16][32];          // per-wave P transpose tile

  const int tid  = threadIdx.x;
  const int lane = tid & 31;
  const int wave = tid >> 5;

  const int bh   = blockIdx.x >> 4;           // 0..7  (b*4 + h)
  const int qblk = blockIdx.x & 15;           // 16 blocks of 128 queries
  const int b = bh >> 2, h = bh & 3;
  const int q0 = qblk * 128 + wave * 16;
  const int rowQ0 = b * SEQ + q0;
  const int bRow = b * SEQ;
  const size_t vglob = ((size_t)b * NHEADS + h) * 64;   // Vt feature-row base

  const int n = lane & 15, half = lane >> 4;
  const int qrow_base = q0 + 8 * half;

  const float sig = (h == 0) ? 5.f : (h == 1) ? 10.f : (h == 2) ? 20.f : 40.f;
  const float is2 = 0.5f / (sig * sig);

  // Q tile A operands (resident for the whole k loop)
  v16bf qa0 = load_a16x32(qp, F_SIZE, rowQ0, h * 64 + 0, lane);
  v16bf qa1 = load_a16x32(qp, F_SIZE, rowQ0, h * 64 + 32, lane);

  // Stage one 8KB K/V tile: waves 0-3 stage K rows, waves 4-7 stage V rows.
  // Each thread moves 32B => exactly 2 async instructions per wave.
  auto stage = [&](int kb, int buf) {
    const uint32_t base = (uint32_t)(uintptr_t)(&kvbuf[buf][0]);
    const int o = tid * 32;                   // byte offset within tile
    if (o < 4096) {                           // K part: [key 0..31][feat 0..63]
      const int row = o >> 7;
      const int col = (o & 127) >> 1;
      const bf16_t* src = kp + (size_t)(bRow + kb + row) * F_SIZE + h * 64 + col;
      async_copy16(base + o,      src);
      async_copy16(base + o + 16, src + 8);
    } else {                                  // V part: [feat 0..63][tok 0..31]
      const int o2 = o - 4096;
      const int fr = o2 >> 6;
      const int tk = (o2 & 63) >> 1;
      const bf16_t* src = vt + (vglob + fr) * SEQ + kb + tk;
      async_copy16(base + o,      src);
      async_copy16(base + o + 16, src + 8);
    }
  };

  const v8f vzero = {0.f, 0.f, 0.f, 0.f, 0.f, 0.f, 0.f, 0.f};
  v8f o0 = vzero, o1 = vzero, o2a = vzero, o3 = vzero;
  float mrow[8], lrow[8];
#pragma unroll
  for (int v = 0; v < 8; ++v) { mrow[v] = -1e30f; lrow[v] = 0.f; }

  stage(0, 0);

  const int NSTEP = SEQ / 32;                 // 64
#pragma unroll 1
  for (int i = 0; i < NSTEP; ++i) {
    const int kb = i * 32;
    if (i + 1 < NSTEP) {
      stage(kb + 32, (i + 1) & 1);            // prefetch next tile
      asm volatile("s_wait_asynccnt 0x2" ::: "memory");  // current tile landed
    } else {
      asm volatile("s_wait_asynccnt 0x0" ::: "memory");
    }
    __syncthreads();                          // publish all waves' staged data

    const bf16_t* kt  = &kvbuf[i & 1][0];     // [32][64]
    const bf16_t* vtl = &kvbuf[i & 1][2048];  // [64][32]

    // --- scores: S[16q x 32k] = Q * K^T (B operands from LDS: ds_load_b128)
    v8f s0 = vzero, s1 = vzero;
    s0 = wmma_bf16(qa0, load_b32x16(kt, 64, 0,  0,  lane), s0);
    s0 = wmma_bf16(qa1, load_b32x16(kt, 64, 0,  32, lane), s0);
    s1 = wmma_bf16(qa0, load_b32x16(kt, 64, 16, 0,  lane), s1);
    s1 = wmma_bf16(qa1, load_b32x16(kt, 64, 16, 32, lane), s1);

    // --- scale + causal Gaussian bias + online softmax
#pragma unroll
    for (int v = 0; v < 8; ++v) {
      const int q = qrow_base + v;
      const int k0i = kb + n, k1i = kb + 16 + n;
      float d0 = (float)(q - k0i), d1 = (float)(q - k1i);
      float b0 = (q >= k0i) ? __expf(-d0 * d0 * is2) : 0.f;
      float b1 = (q >= k1i) ? __expf(-d1 * d1 * is2) : 0.f;
      float x0 = s0[v] * 0.125f + b0;
      float x1 = s1[v] * 0.125f + b1;
      float mx = fmaxf(x0, x1);
#pragma unroll
      for (int off = 8; off; off >>= 1) mx = fmaxf(mx, __shfl_xor(mx, off, 16));
      float mnew = fmaxf(mrow[v], mx);
      float alpha = __expf(mrow[v] - mnew);
      mrow[v] = mnew;
      float p0 = __expf(x0 - mnew);
      float p1 = __expf(x1 - mnew);
      float rs = p0 + p1;
#pragma unroll
      for (int off = 8; off; off >>= 1) rs += __shfl_xor(rs, off, 16);
      lrow[v] = lrow[v] * alpha + rs;
      o0[v] *= alpha; o1[v] *= alpha; o2a[v] *= alpha; o3[v] *= alpha;
      pbuf[wave][v + 8 * half][n]      = (bf16_t)p0;
      pbuf[wave][v + 8 * half][n + 16] = (bf16_t)p1;
    }
    // --- read P back as A operand (same-wave LDS ops are in-order)
    v16bf pa;
    {
      const bf16_t* r = &pbuf[wave][n][half * 8];
      v8bf lo = *reinterpret_cast<const v8bf*>(r);
      v8bf hi = *reinterpret_cast<const v8bf*>(r + 16);
      pa = bf_cat(lo, hi);
    }
    // --- context += P * V (V tile from LDS)
    o0  = wmma_bf16(pa, load_b32x16(vtl, 32, 0,  0, lane), o0);
    o1  = wmma_bf16(pa, load_b32x16(vtl, 32, 16, 0, lane), o1);
    o2a = wmma_bf16(pa, load_b32x16(vtl, 32, 32, 0, lane), o2a);
    o3  = wmma_bf16(pa, load_b32x16(vtl, 32, 48, 0, lane), o3);

    __syncthreads();  // all waves done reading buf[i&1] before it is re-staged
  }

  // --- normalize, store context tile (ctx[token][h*64+feat], bf16)
#pragma unroll
  for (int v = 0; v < 8; ++v) {
    float inv = 1.0f / lrow[v];
    size_t rr = (size_t)(rowQ0 + v + 8 * half) * F_SIZE + h * 64;
    ctx[rr + 0  + n] = (bf16_t)(o0[v]  * inv);
    ctx[rr + 16 + n] = (bf16_t)(o1[v]  * inv);
    ctx[rr + 32 + n] = (bf16_t)(o2a[v] * inv);
    ctx[rr + 48 + n] = (bf16_t)(o3[v]  * inv);
  }
}

// ---------------------------------------------------------------------------
// Kernel 4: out = LayerNorm(ctx @ W_fc + residual). One block = 16 rows.
__global__ void fc_ln_kernel(const bf16_t* __restrict__ ctx,
                             const bf16_t* __restrict__ wfcT,
                             const float* __restrict__ resid,
                             float* __restrict__ out) {
  __shared__ float buf[16][F_SIZE];
  const int lane = threadIdx.x & 31;
  const int wave = threadIdx.x >> 5;
  const int r0 = blockIdx.x * 16;
  const int n = lane & 15, half = lane >> 4;

#pragma unroll
  for (int t = 0; t < 2; ++t) {
    const int col0 = (wave * 2 + t) * 16;
    v8f acc = {0.f, 0.f, 0.f, 0.f, 0.f, 0.f, 0.f, 0.f};
#pragma unroll
    for (int kc = 0; kc < 8; ++kc) {
      v16bf a = load_a16x32(ctx, F_SIZE, r0, kc * 32, lane);
      v16bf b = load_b32x16(wfcT, F_SIZE, col0, kc * 32, lane);
      acc = wmma_bf16(a, b, acc);
    }
#pragma unroll
    for (int v = 0; v < 8; ++v) {
      int rr = v + 8 * half;
      int cc = col0 + n;
      buf[rr][cc] = acc[v] + resid[(size_t)(r0 + rr) * F_SIZE + cc];
    }
  }
  __syncthreads();

  const int row = threadIdx.x >> 4;
  const int seg = threadIdx.x & 15;
  float s = 0.f, s2 = 0.f;
#pragma unroll
  for (int j = 0; j < 16; ++j) {
    float x = buf[row][seg * 16 + j];
    s += x; s2 += x * x;
  }
#pragma unroll
  for (int off = 8; off; off >>= 1) {
    s  += __shfl_xor(s,  off, 16);
    s2 += __shfl_xor(s2, off, 16);
  }
  float mean = s * (1.f / F_SIZE);
  float var  = s2 * (1.f / F_SIZE) - mean * mean;
  float rstd = rsqrtf(var + 1e-5f);
#pragma unroll
  for (int j = 0; j < 16; ++j) {
    out[(size_t)(r0 + row) * F_SIZE + seg * 16 + j] =
        (buf[row][seg * 16 + j] - mean) * rstd;
  }
}

// ---------------------------------------------------------------------------
extern "C" void kernel_launch(void* const* d_in, const int* in_sizes, int n_in,
                              void* d_out, int out_size, void* d_ws, size_t ws_size,
                              hipStream_t stream) {
  (void)in_sizes; (void)n_in; (void)out_size; (void)ws_size;
  const float* inQ = (const float*)d_in[0];
  const float* inK = (const float*)d_in[1];
  const float* inV = (const float*)d_in[2];
  const float* wQ  = (const float*)d_in[3];
  const float* wK  = (const float*)d_in[4];
  const float* wV  = (const float*)d_in[5];
  const float* wFC = (const float*)d_in[6];
  float* out = (float*)d_out;

  char* ws = (char*)d_ws;
  size_t off = 0;
  auto take = [&](size_t bytes) -> void* {
    void* p = ws + off;
    off += (bytes + 255) & ~(size_t)255;
    return p;
  };
  const size_t XB = (size_t)NROWS * F_SIZE;
  const size_t WB = (size_t)F_SIZE * F_SIZE;
  bf16_t* xqb  = (bf16_t*)take(XB * 2);
  bf16_t* xkb  = (bf16_t*)take(XB * 2);
  bf16_t* xvb  = (bf16_t*)take(XB * 2);
  bf16_t* wqT  = (bf16_t*)take(WB * 2);
  bf16_t* wkT  = (bf16_t*)take(WB * 2);
  bf16_t* wvT  = (bf16_t*)take(WB * 2);
  bf16_t* wfcT = (bf16_t*)take(WB * 2);
  bf16_t* qp   = (bf16_t*)take(XB * 2);
  bf16_t* kp   = (bf16_t*)take(XB * 2);
  bf16_t* vt   = (bf16_t*)take(XB * 2);
  bf16_t* ctx  = (bf16_t*)take(XB * 2);

  cvt_inputs_kernel<<<(int)(XB / 256), 256, 0, stream>>>(inQ, inK, inV,
                                                         xqb, xkb, xvb, (int)XB);
  cvt_wt_kernel<<<(int)(WB / 256), 256, 0, stream>>>(wQ,  wqT,  F_SIZE, F_SIZE);
  cvt_wt_kernel<<<(int)(WB / 256), 256, 0, stream>>>(wK,  wkT,  F_SIZE, F_SIZE);
  cvt_wt_kernel<<<(int)(WB / 256), 256, 0, stream>>>(wV,  wvT,  F_SIZE, F_SIZE);
  cvt_wt_kernel<<<(int)(WB / 256), 256, 0, stream>>>(wFC, wfcT, F_SIZE, F_SIZE);

  proj_kernel<<<dim3(512, 3), 256, 0, stream>>>(xqb, xkb, xvb, wqT, wkT, wvT,
                                                qp, kp, vt);

  attn_kernel<<<128, 256, 0, stream>>>(qp, kp, vt, ctx);

  fc_ln_kernel<<<256, 256, 0, stream>>>(ctx, wfcT, inQ, out);
}
